// GNNTransformerDetectorV10_89781996356281
// MI455X (gfx1250) — matrix operand order repaired and checked by reading, not compiled, not run on hardware
//
#include <hip/hip_runtime.h>
#include <hip/hip_bf16.h>
#include <math.h>

typedef __attribute__((ext_vector_type(16))) _Float16 v16h;
typedef __attribute__((ext_vector_type(8)))  _Float16 v8h;
typedef __attribute__((ext_vector_type(8)))  float    v8f;

#define Bc 128
#define Lc 64
#define Kc 32
#define Nc 4
#define HDc 128
#define NHc 4
#define DHc 32
#define BLKROWS (Bc*Lc*Kc)   /* 262144 node rows */
#define BKROWS  (Bc*Kc)      /* 4096 user rows  */
#define EPSc 1e-5f

// ---------------------------------------------------------------------------
// Weight pack: f32 W[KIN x NOUT] -> f16 fragments in WMMA B-layout.
// frag f = nt*NK + kt ; lane l holds col nt*16+(l&15), element e holds
// K = kt*32 + (l>>4)*8 + e (e<8), kt*32 + 16 + (l>>4)*8 + (e-8) (e>=8).  OOB entries zero-filled.
// ---------------------------------------------------------------------------
__global__ void pack_weights_kernel(const float* __restrict__ W, _Float16* __restrict__ out,
                                    int KIN, int NOUT)
{
    int NK = (KIN + 31) / 32;
    int NT = (NOUT + 15) / 16;
    int total = NT * NK * 32 * 16;
    int idx = blockIdx.x * blockDim.x + threadIdx.x;
    if (idx >= total) return;
    int e    = idx & 15;
    int lane = (idx >> 4) & 31;
    int frag = idx >> 9;
    int kt = frag % NK, nt = frag / NK;
    int col = nt * 16 + (lane & 15);
    int kk  = kt * 32 + (lane >> 4) * 8 + (e & 7) + (e >> 3) * 16;
    float f = (kk < KIN && col < NOUT) ? W[(size_t)kk * NOUT + col] : 0.0f;
    out[idx] = (_Float16)f;
}

// ---- fragment load helpers ------------------------------------------------
__device__ __forceinline__ v16h loadA16(const float* base)
{
    float4 x0 = *(const float4*)(base);
    float4 x1 = *(const float4*)(base + 4);
    float4 y0 = *(const float4*)(base + 16);
    float4 y1 = *(const float4*)(base + 20);
    v16h a;
    a[0]=(_Float16)x0.x; a[1]=(_Float16)x0.y; a[2]=(_Float16)x0.z; a[3]=(_Float16)x0.w;
    a[4]=(_Float16)x1.x; a[5]=(_Float16)x1.y; a[6]=(_Float16)x1.z; a[7]=(_Float16)x1.w;
    a[8]=(_Float16)y0.x; a[9]=(_Float16)y0.y; a[10]=(_Float16)y0.z; a[11]=(_Float16)y0.w;
    a[12]=(_Float16)y1.x; a[13]=(_Float16)y1.y; a[14]=(_Float16)y1.z; a[15]=(_Float16)y1.w;
    return a;
}

__device__ __forceinline__ v16h loadB16(const _Float16* p)
{
    v8h lo = *(const v8h*)(p);
    v8h hi = *(const v8h*)(p + 8);
    return __builtin_shufflevector(lo, hi, 0,1,2,3,4,5,6,7,8,9,10,11,12,13,14,15);
}

__device__ __forceinline__ v16h loadH16(const _Float16* hrow, int off)
{
    v8h lo = *(const v8h*)(hrow + off);
    v8h hi = *(const v8h*)(hrow + off + 16);
    return __builtin_shufflevector(lo, hi, 0,1,2,3,4,5,6,7,8,9,10,11,12,13,14,15);
}

// ---------------------------------------------------------------------------
// Single fused MLP layer (used for qkv / out-proj / head).
// ---------------------------------------------------------------------------
template<int KIN, int NOUT, bool RELU, bool RESID>
__global__ __launch_bounds__(256)
void mlp_wmma_kernel(const float* __restrict__ A, int lda,
                     const _Float16* __restrict__ Wp,
                     const float* __restrict__ bias,
                     float* __restrict__ C, int ldc,
                     const float* __restrict__ R)
{
    constexpr int NK = (KIN + 31) / 32;
    constexpr int NT = (NOUT + 15) / 16;
    const int lane  = threadIdx.x & 31;
    const int wave  = threadIdx.x >> 5;
    const int mBase = (blockIdx.x * 8 + wave) * 16;
    const int row   = mBase + (lane & 15);
    const int grp   = lane >> 4;

    v16h afrag[NK];
    #pragma unroll
    for (int kt = 0; kt < NK; ++kt) {
        if constexpr (KIN % 32 == 0) {
            afrag[kt] = loadA16(A + (size_t)row * lda + kt * 32 + grp * 8);
        } else {
            v16h a;
            #pragma unroll
            for (int j = 0; j < 8; ++j) {
                int k1 = kt * 32 + grp * 8 + j;
                int k2 = k1 + 16;
                float f1 = A[(size_t)row * lda + (k1 < KIN ? k1 : 0)];
                float f2 = A[(size_t)row * lda + (k2 < KIN ? k2 : 0)];
                a[j]     = (_Float16)((k1 < KIN) ? f1 : 0.0f);
                a[j + 8] = (_Float16)((k2 < KIN) ? f2 : 0.0f);
            }
            afrag[kt] = a;
        }
    }
    for (int nt = 0; nt < NT; ++nt) {
        v8f acc = {};
        const _Float16* wlane = Wp + ((size_t)(nt * NK) * 32 + lane) * 16;
        #pragma unroll
        for (int kt = 0; kt < NK; ++kt)
            acc = __builtin_amdgcn_wmma_f32_16x16x32_f16(
                false, afrag[kt], false, loadB16(wlane + (size_t)kt * 512),
                (short)0, acc, false, false);
        const int col = nt * 16 + (lane & 15);
        float bv = bias[(NOUT % 16 == 0) ? col : (col < NOUT ? col : 0)];
        if ((NOUT % 16 == 0) || col < NOUT) {
            #pragma unroll
            for (int v = 0; v < 8; ++v) {
                size_t off = (size_t)(mBase + grp * 8 + v) * ldc + col;
                float val = acc[v] + bv;
                if (RELU)  val = fmaxf(val, 0.0f);
                if (RESID) val += R[off];
                C[off] = val;
            }
        }
    }
}

// ---------------------------------------------------------------------------
// Fused two-layer MLP: C = [BN]( relu(A@W1+b1) @ W2 + b2 [+R] ).
// Hidden tile staged in wave-private LDS (f16) to transpose D->A layout;
// same-wave DS ops are in-order so no barriers are required.
// 4 waves / 128 threads per block, 64 rows per block.
// ---------------------------------------------------------------------------
template<int KIN, int KHID, int NOUT, bool BN, bool RESID>
__global__ __launch_bounds__(128)
void mlp2_wmma_kernel(const float* __restrict__ A, int lda,
                      const _Float16* __restrict__ W1p, const float* __restrict__ b1,
                      const _Float16* __restrict__ W2p, const float* __restrict__ b2,
                      float* __restrict__ C, int ldc,
                      const float* __restrict__ R,
                      const float* __restrict__ bng, const float* __restrict__ bnb,
                      const float* __restrict__ bnrm, const float* __restrict__ bnrv)
{
    constexpr int NK1 = (KIN + 31) / 32;
    constexpr int NT1 = KHID / 16;
    constexpr int NK2 = KHID / 32;
    constexpr int NT2 = (NOUT + 15) / 16;

    __shared__ _Float16 hlds[4 * 16 * KHID];

    const int lane  = threadIdx.x & 31;
    const int wave  = threadIdx.x >> 5;
    const int mBase = (blockIdx.x * 4 + wave) * 16;
    const int row   = mBase + (lane & 15);
    const int grp   = lane >> 4;
    _Float16* hbase = hlds + wave * 16 * KHID;

    // ---- layer 1: A @ W1, relu, stage into LDS (A layout for layer 2) ----
    v16h afrag[NK1];
    #pragma unroll
    for (int kt = 0; kt < NK1; ++kt) {
        if constexpr (KIN % 32 == 0) {
            afrag[kt] = loadA16(A + (size_t)row * lda + kt * 32 + grp * 8);
        } else {
            v16h a;
            #pragma unroll
            for (int j = 0; j < 8; ++j) {
                int k1 = kt * 32 + grp * 8 + j;
                int k2 = k1 + 16;
                float f1 = A[(size_t)row * lda + (k1 < KIN ? k1 : 0)];
                float f2 = A[(size_t)row * lda + (k2 < KIN ? k2 : 0)];
                a[j]     = (_Float16)((k1 < KIN) ? f1 : 0.0f);
                a[j + 8] = (_Float16)((k2 < KIN) ? f2 : 0.0f);
            }
            afrag[kt] = a;
        }
    }
    #pragma unroll
    for (int nt = 0; nt < NT1; ++nt) {
        v8f acc = {};
        const _Float16* wlane = W1p + ((size_t)(nt * NK1) * 32 + lane) * 16;
        #pragma unroll
        for (int kt = 0; kt < NK1; ++kt)
            acc = __builtin_amdgcn_wmma_f32_16x16x32_f16(
                false, afrag[kt], false, loadB16(wlane + (size_t)kt * 512),
                (short)0, acc, false, false);
        const int col = nt * 16 + (lane & 15);
        float bv = b1[col];
        #pragma unroll
        for (int v = 0; v < 8; ++v)
            hbase[(grp * 8 + v) * KHID + col] = (_Float16)fmaxf(acc[v] + bv, 0.0f);
    }

    // ---- layer 2: hidden @ W2 (A fragments read from LDS) ----------------
    const _Float16* hrow = hbase + (lane & 15) * KHID;
    v16h hfrag[NK2];
    #pragma unroll
    for (int kt = 0; kt < NK2; ++kt)
        hfrag[kt] = loadH16(hrow, kt * 32 + grp * 8);

    for (int nt = 0; nt < NT2; ++nt) {
        v8f acc = {};
        const _Float16* wlane = W2p + ((size_t)(nt * NK2) * 32 + lane) * 16;
        #pragma unroll
        for (int kt = 0; kt < NK2; ++kt)
            acc = __builtin_amdgcn_wmma_f32_16x16x32_f16(
                false, hfrag[kt], false, loadB16(wlane + (size_t)kt * 512),
                (short)0, acc, false, false);
        const int col = nt * 16 + (lane & 15);
        float bv = b2[(NOUT % 16 == 0) ? col : (col < NOUT ? col : 0)];
        float bscale = 1.f, bshift = 0.f;
        if constexpr (BN) {
            int cc = (NOUT % 16 == 0) ? col : (col < NOUT ? col : 0);
            bscale = rsqrtf(bnrv[cc] + EPSc) * bng[cc];
            bshift = bnb[cc] - bnrm[cc] * bscale;
        }
        if ((NOUT % 16 == 0) || col < NOUT) {
            #pragma unroll
            for (int v = 0; v < 8; ++v) {
                size_t off = (size_t)(mBase + grp * 8 + v) * ldc + col;
                float val = acc[v] + bv;
                if (BN)    val = val * bscale + bshift;
                if (RESID) val += R[off];
                C[off] = val;
            }
        }
    }
}

// ---------------------------------------------------------------------------
// GNN update, fully fused: upd = MLP2([h | mean_bcast]) with K=256 fragments
// sourced directly from h (K 0..127) and the per-(b,k) mean row (K 128..255),
// eliminating the cat256 tensor entirely.
// ---------------------------------------------------------------------------
__global__ __launch_bounds__(128)
void gnn_upd_kernel(const float* __restrict__ h, const float* __restrict__ mean,
                    const _Float16* __restrict__ W1p, const float* __restrict__ b1,
                    const _Float16* __restrict__ W2p, const float* __restrict__ b2,
                    float* __restrict__ C)
{
    constexpr int KHID = 128;
    __shared__ _Float16 hlds[4 * 16 * KHID];

    const int lane  = threadIdx.x & 31;
    const int wave  = threadIdx.x >> 5;
    const int mBase = (blockIdx.x * 4 + wave) * 16;
    const int row   = mBase + (lane & 15);
    const int grp   = lane >> 4;
    const int mrow  = (row >> 11) * Kc + (row & (Kc - 1));   // (b, k) mean row
    _Float16* hbase = hlds + wave * 16 * KHID;

    v16h afrag[8];
    #pragma unroll
    for (int kt = 0; kt < 4; ++kt)
        afrag[kt] = loadA16(h + (size_t)row * 128 + kt * 32 + grp * 8);
    #pragma unroll
    for (int kt = 0; kt < 4; ++kt)
        afrag[4 + kt] = loadA16(mean + (size_t)mrow * 128 + kt * 32 + grp * 8);

    #pragma unroll
    for (int nt = 0; nt < 8; ++nt) {
        v8f acc = {};
        const _Float16* wlane = W1p + ((size_t)(nt * 8) * 32 + lane) * 16;
        #pragma unroll
        for (int kt = 0; kt < 8; ++kt)
            acc = __builtin_amdgcn_wmma_f32_16x16x32_f16(
                false, afrag[kt], false, loadB16(wlane + (size_t)kt * 512),
                (short)0, acc, false, false);
        const int col = nt * 16 + (lane & 15);
        float bv = b1[col];
        #pragma unroll
        for (int v = 0; v < 8; ++v)
            hbase[(grp * 8 + v) * KHID + col] = (_Float16)fmaxf(acc[v] + bv, 0.0f);
    }

    const _Float16* hrow = hbase + (lane & 15) * KHID;
    v16h hfrag[4];
    #pragma unroll
    for (int kt = 0; kt < 4; ++kt)
        hfrag[kt] = loadH16(hrow, kt * 32 + grp * 8);

    #pragma unroll
    for (int nt = 0; nt < 8; ++nt) {
        v8f acc = {};
        const _Float16* wlane = W2p + ((size_t)(nt * 4) * 32 + lane) * 16;
        #pragma unroll
        for (int kt = 0; kt < 4; ++kt)
            acc = __builtin_amdgcn_wmma_f32_16x16x32_f16(
                false, hfrag[kt], false, loadB16(wlane + (size_t)kt * 512),
                (short)0, acc, false, false);
        const int col = nt * 16 + (lane & 15);
        float bv = b2[col];
        #pragma unroll
        for (int v = 0; v < 8; ++v)
            C[(size_t)(mBase + grp * 8 + v) * 128 + col] = acc[v] + bv;
    }
}

// --------------------------- elementwise helpers ---------------------------

__global__ void gather_hflat_kernel(const float* __restrict__ Hq, float* __restrict__ out)
{
    int idx = blockIdx.x * blockDim.x + threadIdx.x;
    if (idx >= BLKROWS * 8) return;
    int c = idx & 7, row = idx >> 3;
    int k = row % Kc, bl = row / Kc;
    int n = c >> 1, comp = c & 1;
    out[idx] = Hq[((((size_t)bl * Nc) + n) * Kc + k) * 2 + comp];
}

__global__ void hpow_kernel(const float* __restrict__ hflat, float* __restrict__ hpow)
{
    int idx = blockIdx.x * blockDim.x + threadIdx.x;
    if (idx >= BLKROWS) return;
    float s = 0.f;
    #pragma unroll
    for (int j = 0; j < 8; ++j) { float v = hflat[(size_t)idx * 8 + j]; s += v * v; }
    hpow[idx] = s;
}

__global__ void sumpow_kernel(const float* __restrict__ hpow, float* __restrict__ sump)
{
    int bl = blockIdx.x * blockDim.x + threadIdx.x;
    if (bl >= Bc * Lc) return;
    float s = 0.f;
    for (int k = 0; k < Kc; ++k) s += hpow[(size_t)bl * Kc + k];
    sump[bl] = s;
}

__global__ void tail_kernel(const float* __restrict__ bw, const float* __restrict__ snr,
                            const float* __restrict__ hpow, const float* __restrict__ sump,
                            float* __restrict__ cat)
{
    int idx = blockIdx.x * blockDim.x + threadIdx.x;
    if (idx >= BLKROWS) return;
    int bl = idx / Kc;
    size_t base = (size_t)idx * 70;
    cat[base + 64] = bw[(size_t)idx * 3 + 0];
    cat[base + 65] = bw[(size_t)idx * 3 + 1];
    cat[base + 66] = bw[(size_t)idx * 3 + 2];
    float hp = hpow[idx];
    cat[base + 67] = log1pf(hp);
    cat[base + 68] = log1pf(sump[bl] - hp);
    cat[base + 69] = snr[idx];
}

__global__ void meanL_kernel(const float* __restrict__ msg, float* __restrict__ out)
{
    int idx = blockIdx.x * blockDim.x + threadIdx.x;
    if (idx >= BKROWS * HDc) return;
    int c = idx & (HDc - 1);
    int bk = idx >> 7;
    int b = bk / Kc, k = bk % Kc;
    float s = 0.f;
    for (int l = 0; l < Lc; ++l)
        s += msg[(((size_t)(b * Lc + l)) * Kc + k) * HDc + c];
    out[idx] = s * (1.f / Lc);
}

__global__ void resln_kernel(float* __restrict__ h, const float* __restrict__ upd,
                             const float* __restrict__ g, const float* __restrict__ bb,
                             int rows)
{
    int gid  = blockIdx.x * blockDim.x + threadIdx.x;
    int wid  = gid >> 5;
    int lane = gid & 31;
    if (wid >= rows) return;
    float v[4]; float s = 0.f;
    #pragma unroll
    for (int i = 0; i < 4; ++i) {
        int c = lane * 4 + i;
        float t = h[(size_t)wid * HDc + c] + upd[(size_t)wid * HDc + c];
        v[i] = t; s += t;
    }
    #pragma unroll
    for (int m = 16; m > 0; m >>= 1) s += __shfl_xor(s, m, 32);
    float mean = s * (1.f / HDc);
    float var = 0.f;
    #pragma unroll
    for (int i = 0; i < 4; ++i) { float d = v[i] - mean; var += d * d; }
    #pragma unroll
    for (int m = 16; m > 0; m >>= 1) var += __shfl_xor(var, m, 32);
    float inv = rsqrtf(var * (1.f / HDc) + EPSc);
    #pragma unroll
    for (int i = 0; i < 4; ++i) {
        int c = lane * 4 + i;
        h[(size_t)wid * HDc + c] = (v[i] - mean) * inv * g[c] + bb[c];
    }
}

__global__ void ln_kernel(const float* __restrict__ src, float* __restrict__ dst,
                          const float* __restrict__ g, const float* __restrict__ bb,
                          int rows)
{
    int gid  = blockIdx.x * blockDim.x + threadIdx.x;
    int wid  = gid >> 5;
    int lane = gid & 31;
    if (wid >= rows) return;
    float v[4]; float s = 0.f;
    #pragma unroll
    for (int i = 0; i < 4; ++i) {
        int c = lane * 4 + i;
        float t = src[(size_t)wid * HDc + c];
        v[i] = t; s += t;
    }
    #pragma unroll
    for (int m = 16; m > 0; m >>= 1) s += __shfl_xor(s, m, 32);
    float mean = s * (1.f / HDc);
    float var = 0.f;
    #pragma unroll
    for (int i = 0; i < 4; ++i) { float d = v[i] - mean; var += d * d; }
    #pragma unroll
    for (int m = 16; m > 0; m >>= 1) var += __shfl_xor(var, m, 32);
    float inv = rsqrtf(var * (1.f / HDc) + EPSc);
    #pragma unroll
    for (int i = 0; i < 4; ++i) {
        int c = lane * 4 + i;
        dst[(size_t)wid * HDc + c] = (v[i] - mean) * inv * g[c] + bb[c];
    }
}

__global__ void softmaxL_kernel(const float* __restrict__ logit, float* __restrict__ w)
{
    __shared__ float red[64];
    int bk = blockIdx.x;
    int b = bk / Kc, k = bk % Kc;
    int l = threadIdx.x;
    size_t idx = ((size_t)(b * Lc + l)) * Kc + k;
    float v = logit[idx];
    red[l] = v; __syncthreads();
    for (int s = 32; s > 0; s >>= 1) { if (l < s) red[l] = fmaxf(red[l], red[l + s]); __syncthreads(); }
    float mx = red[0]; __syncthreads();
    float e = expf(v - mx);
    red[l] = e; __syncthreads();
    for (int s = 32; s > 0; s >>= 1) { if (l < s) red[l] += red[l + s]; __syncthreads(); }
    w[idx] = e / red[0];
}

__global__ void mrcfeat_kernel(const float* __restrict__ Hq, const float* __restrict__ yq,
                               float* __restrict__ out)
{
    int bk = blockIdx.x * blockDim.x + threadIdx.x;
    if (bk >= BKROWS) return;
    int b = bk / Kc, k = bk % Kc;
    float sr = 0.f, si = 0.f;
    for (int ln = 0; ln < Lc * Nc; ++ln) {
        int l = ln / Nc, n = ln % Nc;
        size_t hb = ((((size_t)b * Lc + l) * Nc + n) * Kc + k) * 2;
        size_t yb = (((size_t)b * Lc + l) * Nc + n) * 2;
        float hr = Hq[hb], hi = Hq[hb + 1];
        float yr = yq[yb], yi = yq[yb + 1];
        sr += hr * yr + hi * yi;
        si += hr * yi - hi * yr;
    }
    out[(size_t)bk * 2]     = sr;
    out[(size_t)bk * 2 + 1] = si;
}

__global__ void pool_kernel(const float* __restrict__ h, const float* __restrict__ aw,
                            const float* __restrict__ mrce, float* __restrict__ x)
{
    int idx = blockIdx.x * blockDim.x + threadIdx.x;
    if (idx >= BKROWS * HDc) return;
    int c = idx & (HDc - 1);
    int bk = idx >> 7;
    int b = bk / Kc, k = bk % Kc;
    float s = 0.f;
    for (int l = 0; l < Lc; ++l) {
        size_t r = ((size_t)(b * Lc + l)) * Kc + k;
        s += h[r * HDc + c] * aw[r];
    }
    x[idx] = s + mrce[idx];
}

__global__ void mha_kernel(const float* __restrict__ qkv, float* __restrict__ o)
{
    int bh = blockIdx.x;
    int b = bh / NHc, hh = bh % NHc;
    int lane = threadIdx.x;
    const float* base = qkv + (size_t)b * Kc * (3 * HDc);
    float q[DHc];
    #pragma unroll
    for (int d = 0; d < DHc; ++d) q[d] = base[(size_t)lane * 384 + hh * DHc + d];
    float sc[Kc]; float mx = -1e30f;
    const float scale = 0.17677669529663689f;
    for (int j = 0; j < Kc; ++j) {
        float s = 0.f;
        #pragma unroll
        for (int d = 0; d < DHc; ++d) s += q[d] * base[(size_t)j * 384 + HDc + hh * DHc + d];
        sc[j] = s * scale; mx = fmaxf(mx, sc[j]);
    }
    float sum = 0.f;
    for (int j = 0; j < Kc; ++j) { sc[j] = expf(sc[j] - mx); sum += sc[j]; }
    float inv = 1.f / sum;
    for (int d = 0; d < DHc; ++d) {
        float acc = 0.f;
        for (int j = 0; j < Kc; ++j) acc += sc[j] * base[(size_t)j * 384 + 2 * HDc + hh * DHc + d];
        o[((size_t)b * Kc + lane) * HDc + hh * DHc + d] = acc * inv;
    }
}

__global__ void final_kernel(const float* __restrict__ shat, const float* __restrict__ heado,
                             const float* __restrict__ alpha, const float* __restrict__ bscale,
                             float* __restrict__ out)
{
    int bk = blockIdx.x * blockDim.x + threadIdx.x;
    if (bk >= BKROWS) return;
    int b = bk / Kc, k = bk % Kc;
    float ar = 0.f, ai = 0.f;
    for (int l = 0; l < Lc; ++l) {
        size_t base = (((size_t)b * Lc + l) * Kc + k) * 2;
        ar += shat[base]; ai += shat[base + 1];
    }
    ar *= (1.f / Lc); ai *= (1.f / Lc);
    const float c = 0.70710678118654752f;
    const float cx[4] = { c, -c, -c,  c };
    const float cy[4] = { c,  c, -c, -c };
    float a = alpha[0], bs = bscale[0];
    #pragma unroll
    for (int j = 0; j < 4; ++j) {
        float dx = ar - cx[j], dy = ai - cy[j];
        out[(size_t)bk * 4 + j] = -(dx * dx + dy * dy) * bs + a * heado[(size_t)bk * 4 + j];
    }
}

// --------------------------- param index map (sorted pytree order) ---------
static inline int cdiv(int a, int b) { return (a + b - 1) / b; }
static inline size_t packElems(int kin, int nout)
{ return (size_t)((kin + 31) / 32) * ((nout + 15) / 16) * 512; }

extern "C" void kernel_launch(void* const* d_in, const int* in_sizes, int n_in,
                              void* d_out, int out_size, void* d_ws, size_t ws_size,
                              hipStream_t stream)
{
    (void)in_sizes; (void)n_in; (void)out_size; (void)ws_size;
    const float* shat = (const float*)d_in[0];
    const float* Hq   = (const float*)d_in[1];
    const float* yq   = (const float*)d_in[2];
    const float* bw   = (const float*)d_in[3];
    const float* snr  = (const float*)d_in[4];
#define P(i) ((const float*)d_in[i])

    float* ws   = (float*)d_ws;
    float* cat  = ws;                              // BLK*256 (combined70 lives here)
    float* hbuf = cat  + (size_t)BLKROWS * 256;
    float* t1   = hbuf + (size_t)BLKROWS * 128;    // reserved
    float* t2   = t1   + (size_t)BLKROWS * 128;    // hflat / messages / upd
    float* hpow = t2   + (size_t)BLKROWS * 128;
    float* alog = hpow + BLKROWS;
    float* aw   = alog + BLKROWS;
    float* sm   = aw   + BLKROWS;
    float* sump    = sm;
    float* meanb   = sump + 8192;
    float* mrcfeat = meanb + 524288;
    float* mrch    = mrcfeat + 8192;               // reserved
    float* mrce    = mrch + 262144;
    float* xbuf    = mrce + 524288;
    float* xn      = xbuf + 524288;
    float* qkvb    = xn + 524288;
    float* atto    = qkvb + 1572864;
    float* ffh     = atto + 524288;                // reserved
    float* heado   = ffh + 1048576;

    _Float16* wp = (_Float16*)(heado + 16384);
    const int TB = 256;
    struct WSpec { int idx, kin, nout; };
    const WSpec specs[] = {
        {19, 2, 32},   {21, 32, 32},  {15, 8, 32},   {17, 32, 32},
        {23, 70, 128}, {25, 128, 128},
        {29, 128,128}, {31, 128,128}, {33, 256,128}, {35, 128,128},
        {39, 128,128}, {41, 128,128}, {43, 256,128}, {45, 128,128},
        {49, 128,128}, {51, 128,128}, {53, 256,128}, {55, 128,128},
        {6, 128, 64},  {8, 64, 1},
        {59, 2, 64},   {61, 64, 128},
        {73, 128,384}, {71, 128,128}, {63, 128,256}, {65, 256,128},
        {85, 128,384}, {83, 128,128}, {75, 128,256}, {77, 256,128},
        {57, 128, 4},
    };
    _Float16* wptr[31];
    {
        _Float16* cur = wp;
        for (int i = 0; i < 31; ++i) {
            wptr[i] = cur;
            size_t n = packElems(specs[i].kin, specs[i].nout);
            pack_weights_kernel<<<cdiv((int)n, TB), TB, 0, stream>>>(
                P(specs[i].idx), cur, specs[i].kin, specs[i].nout);
            cur += n;
        }
    }
    enum { W_DM0, W_DM1, W_CH0, W_CH1, W_FU0, W_FU1,
           W_G0M0, W_G0M1, W_G0U0, W_G0U1,
           W_G1M0, W_G1M1, W_G1U0, W_G1U1,
           W_G2M0, W_G2M1, W_G2U0, W_G2U1,
           W_AP0, W_AP1, W_MRC0, W_MRC1,
           W_T0QKV, W_T0OUT, W_T0FF0, W_T0FF1,
           W_T1QKV, W_T1OUT, W_T1FF0, W_T1FF1,
           W_HEAD };

    const int gNode2 = BLKROWS / 64;   // fused: 4 waves * 16 rows
    const int gUser2 = BKROWS / 64;
    const int gUser1 = BKROWS / 128;   // single: 8 waves * 16 rows
    const float* NUL = nullptr;

    // ---- node feature construction ----
    gather_hflat_kernel<<<cdiv(BLKROWS * 8, TB), TB, 0, stream>>>(Hq, t2);
    mlp2_wmma_kernel<2,32,32,false,false><<<gNode2, 128, 0, stream>>>(
        shat, 2, wptr[W_DM0], P(20), wptr[W_DM1], P(22), cat, 70, NUL, NUL, NUL, NUL, NUL);
    mlp2_wmma_kernel<8,32,32,false,false><<<gNode2, 128, 0, stream>>>(
        t2, 8, wptr[W_CH0], P(16), wptr[W_CH1], P(18), cat + 32, 70, NUL, NUL, NUL, NUL, NUL);
    hpow_kernel<<<cdiv(BLKROWS, TB), TB, 0, stream>>>(t2, hpow);
    sumpow_kernel<<<cdiv(Bc * Lc, TB), TB, 0, stream>>>(hpow, sump);
    tail_kernel<<<cdiv(BLKROWS, TB), TB, 0, stream>>>(bw, snr, hpow, sump, cat);
    // fusion MLP with BN folded into the epilogue
    mlp2_wmma_kernel<70,128,128,true,false><<<gNode2, 128, 0, stream>>>(
        cat, 70, wptr[W_FU0], P(24), wptr[W_FU1], P(26), hbuf, 128, NUL,
        P(12), P(11), P(13), P(14));

    // ---- 3 GNN layers ----
    const int gw[3][4] = { {W_G0M0, W_G0M1, W_G0U0, W_G0U1},
                           {W_G1M0, W_G1M1, W_G1U0, W_G1U1},
                           {W_G2M0, W_G2M1, W_G2U0, W_G2U1} };
    for (int layer = 0; layer < 3; ++layer) {
        int g = 27 + 10 * layer;
        mlp2_wmma_kernel<128,128,128,false,false><<<gNode2, 128, 0, stream>>>(
            hbuf, 128, wptr[gw[layer][0]], P(g + 3), wptr[gw[layer][1]], P(g + 5),
            t2, 128, NUL, NUL, NUL, NUL, NUL);
        meanL_kernel<<<cdiv(BKROWS * HDc, TB), TB, 0, stream>>>(t2, meanb);
        gnn_upd_kernel<<<gNode2, 128, 0, stream>>>(
            hbuf, meanb, wptr[gw[layer][2]], P(g + 7), wptr[gw[layer][3]], P(g + 9), t2);
        resln_kernel<<<cdiv(BLKROWS * 32, TB), TB, 0, stream>>>(hbuf, t2, P(g + 1), P(g + 0), BLKROWS);
    }

    // ---- attention pooling + MRC embed ----
    mlp2_wmma_kernel<128,64,1,false,false><<<gNode2, 128, 0, stream>>>(
        hbuf, 128, wptr[W_AP0], P(7), wptr[W_AP1], P(9), alog, 1, NUL, NUL, NUL, NUL, NUL);
    softmaxL_kernel<<<BKROWS, 64, 0, stream>>>(alog, aw);
    mrcfeat_kernel<<<cdiv(BKROWS, TB), TB, 0, stream>>>(Hq, yq, mrcfeat);
    mlp2_wmma_kernel<2,64,128,false,false><<<gUser2, 128, 0, stream>>>(
        mrcfeat, 2, wptr[W_MRC0], P(60), wptr[W_MRC1], P(62), mrce, 128, NUL, NUL, NUL, NUL, NUL);
    pool_kernel<<<cdiv(BKROWS * HDc, TB), TB, 0, stream>>>(hbuf, aw, mrce, xbuf);

    // ---- 2 transformer encoder layers (norm-first, residuals fused) ----
    const int tw[2][4] = { {W_T0QKV, W_T0OUT, W_T0FF0, W_T0FF1},
                           {W_T1QKV, W_T1OUT, W_T1FF0, W_T1FF1} };
    for (int t = 0; t < 2; ++t) {
        int tb = 63 + 12 * t;
        ln_kernel<<<cdiv(BKROWS * 32, TB), TB, 0, stream>>>(xbuf, xn, P(tb + 5), P(tb + 4), BKROWS);
        mlp_wmma_kernel<128,384,false,false><<<gUser1, TB, 0, stream>>>(
            xn, 128, wptr[tw[t][0]], P(tb + 11), qkvb, 384, NUL);
        mha_kernel<<<Bc * NHc, 32, 0, stream>>>(qkvb, atto);
        mlp_wmma_kernel<128,128,false,true><<<gUser1, TB, 0, stream>>>(
            atto, 128, wptr[tw[t][1]], P(tb + 9), xbuf, 128, xbuf);
        ln_kernel<<<cdiv(BKROWS * 32, TB), TB, 0, stream>>>(xbuf, xn, P(tb + 7), P(tb + 6), BKROWS);
        mlp2_wmma_kernel<128,256,128,false,true><<<gUser2, 128, 0, stream>>>(
            xn, 128, wptr[tw[t][2]], P(tb + 1), wptr[tw[t][3]], P(tb + 3),
            xbuf, 128, xbuf, NUL, NUL, NUL, NUL);
    }

    // ---- head + constellation logits ----
    mlp_wmma_kernel<128,4,false,false><<<gUser1, TB, 0, stream>>>(
        xbuf, 128, wptr[W_HEAD], P(58), heado, 4, NUL);
    final_kernel<<<cdiv(BKROWS, TB), TB, 0, stream>>>(shat, heado, P(5), P(10), (float*)d_out);
#undef P
}
